// TransformerCRF_66864050864158
// MI455X (gfx1250) — matrix operand order repaired and checked by reading, not compiled, not run on hardware
//
#include <hip/hip_runtime.h>
#include <hip/hip_bf16.h>

// ---------------------------------------------------------------------------
// TransformerCRF for MI455X (gfx1250): bf16 WMMA GEMMs + flash attention +
// fp32 LayerNorm + CRF log-semiring scan.
// Round 3: fix async-to-LDS builtin arg types — it takes pointers to
// 16-byte int vectors (as1 / as3), not void*.
// ---------------------------------------------------------------------------

// Problem constants (match reference)
constexpr int NB  = 32;      // batch
constexpr int NS  = 512;     // seq len
constexpr int NE  = 512;     // embed dim
constexpr int NDF = 2048;    // ffn hidden
constexpr int NH  = 8;       // heads
constexpr int NL  = 2;       // layers
constexpr int NT  = 12;      // tags
constexpr int TAG_START = 10;
constexpr int TAG_STOP  = 11;
constexpr float NEGV = -10000.0f;

typedef __attribute__((ext_vector_type(16))) __bf16 v16bf;
typedef __attribute__((ext_vector_type(8)))  float  v8f;

union BF16Frag { v16bf v; unsigned int u[8]; };
union F32Acc   { v8f   v; float        f[8]; };

#define AS1 __attribute__((address_space(1)))
#define AS3 __attribute__((address_space(3)))

// 16-byte int vector matching the async builtin's parameter type
typedef int async_v4i __attribute__((vector_size(16)));

#if __has_builtin(__builtin_amdgcn_global_load_async_to_lds_b128) && \
    __has_builtin(__builtin_amdgcn_s_wait_asynccnt)
#define HAS_ASYNC_LDS 1
#else
#define HAS_ASYNC_LDS 0
#endif

#if __has_builtin(__builtin_amdgcn_tensor_load_to_lds) && \
    __has_builtin(__builtin_amdgcn_s_wait_tensorcnt)
#define HAS_TDM 1
typedef __attribute__((ext_vector_type(4))) unsigned int tdm_v4u;
typedef __attribute__((ext_vector_type(8))) int          tdm_v8i;
typedef __attribute__((ext_vector_type(4))) int          tdm_v4i;
#else
#define HAS_TDM 0
#endif

__device__ inline float redSum32(float v) {
  #pragma unroll
  for (int m = 16; m >= 1; m >>= 1) v += __shfl_xor(v, m, 32);
  return v;
}
__device__ inline float redMax16(float v) {
  #pragma unroll
  for (int m = 8; m >= 1; m >>= 1) v = fmaxf(v, __shfl_xor(v, m, 16));
  return v;
}
__device__ inline float redSum16(float v) {
  #pragma unroll
  for (int m = 8; m >= 1; m >>= 1) v += __shfl_xor(v, m, 16);
  return v;
}

// ---------------------------------------------------------------------------
// Embedding gather: x[token][e] = emb[sentence[token]][e]
// ---------------------------------------------------------------------------
__global__ __launch_bounds__(256) void embed_gather(
    const int* __restrict__ sent, const float* __restrict__ emb,
    float* __restrict__ x) {
  const int tok = blockIdx.x;
  const int id  = sent[tok];
  const float* src = emb + (size_t)id * NE;
  float* dst = x + (size_t)tok * NE;
  for (int c = threadIdx.x; c < NE; c += blockDim.x) dst[c] = src[c];
}

// ---------------------------------------------------------------------------
// Generic C = act(A @ W^T + bias).  A:(M,K) row-major (f32 or bf16),
// W:(N,K) row-major f32, C:(M,N) f32 or bf16.
// Block tile 128x64, K-chunk 32, 8 waves, each wave: 32x32 = 4 WMMA tiles.
// bf16 A-tiles are staged by the Tensor Data Mover when available.
// ---------------------------------------------------------------------------
#define BM 128
#define BN 64
#define BK 32
#define LDA 40   // LDS row stride (bf16 elems): 32 data + 8 pad (16dw + 4dw)
#define LDB 40

template <int A_IS_BF16, int OUT_BF16, int RELU>
__global__ __launch_bounds__(256) void gemm_xwT(
    const void* __restrict__ Araw, const float* __restrict__ W,
    const float* __restrict__ bias, void* __restrict__ Craw,
    int M, int N, int K) {
  __shared__ __bf16 As[BM * LDA];
  __shared__ __bf16 Bs[BN * LDB];

  const int tid   = threadIdx.x;
  const int lane  = tid & 31;
  const int wave  = tid >> 5;
  const int wm    = wave >> 1;           // 4x2 wave grid over 128x64 tile
  const int wn    = wave & 1;
  const int lhalf = lane >> 4;
  const int l15   = lane & 15;

  const int mBase = blockIdx.y * BM;
  const int nBase = blockIdx.x * BN;

  F32Acc acc[2][2];
  #pragma unroll
  for (int i = 0; i < 2; i++)
    #pragma unroll
    for (int j = 0; j < 2; j++)
      #pragma unroll
      for (int r = 0; r < 8; r++) acc[i][j].f[r] = 0.0f;

  // staging assignments
  const int ar = tid >> 1, ac = (tid & 1) * 16;   // A: 128 rows x 32, 16/thread
  const int br = tid >> 2, bc = (tid & 3) * 8;    // B: 64 rows  x 32,  8/thread

  for (int kt = 0; kt < K; kt += BK) {
    __syncthreads();
    // ---- stage A tile ----
#if HAS_TDM
    if (A_IS_BF16) {
      // Tensor Data Mover: one wave issues a 2D D# (tile 32x128 bf16,
      // tensor dim0=K stride K, LDS padded 16 dwords -> +4 dwords = LDA 40).
      if (wave == 0) {
        const unsigned long long ga =
            (unsigned long long)(uintptr_t)Araw +
            (((size_t)mBase * (size_t)K + (size_t)kt) << 1);
        const unsigned ldsAddr = (unsigned)(uintptr_t)(AS3 void*)&As[0];
        tdm_v4u g0;
        g0[0] = 1u;                                   // count=1 (valid D#)
        g0[1] = ldsAddr;                              // lds_addr
        g0[2] = (unsigned)ga;                         // global_addr[31:0]
        g0[3] = (unsigned)(ga >> 32) | (2u << 30);    // ga[56:32] | type=2
        tdm_v8i g1;
        g1[0] = (1 << 16) |        // data_size = 2 bytes
                (1 << 20) |        // pad_enable
                (3 << 22) |        // pad_interval: 16 dwords
                (3 << 25);         // pad_amount:   4 dwords
        g1[1] = (int)(((unsigned)K & 0xFFFFu) << 16);             // dim0 lo
        g1[2] = (int)((((unsigned)K >> 16) & 0xFFFFu) |
                      (((unsigned)M & 0xFFFFu) << 16));           // dim0 hi|dim1 lo
        g1[3] = (int)((((unsigned)M >> 16) & 0xFFFFu) |
                      (32u << 16));                               // dim1 hi|tile0=32
        g1[4] = 128;                                              // tile1=128
        g1[5] = (int)(unsigned)K;                                 // dim0_stride lo32
        g1[6] = 0;                                                // stride hi / s1 lo
        g1[7] = 0;
        tdm_v4i gz = {0, 0, 0, 0};
#if defined(__clang_major__) && __clang_major__ >= 23
        tdm_v8i z8 = {0, 0, 0, 0, 0, 0, 0, 0};
        __builtin_amdgcn_tensor_load_to_lds(g0, g1, gz, gz, z8, 0);
#else
        __builtin_amdgcn_tensor_load_to_lds(g0, g1, gz, gz, 0);
#endif
      }
    } else
#endif
    {
      const int gm = mBase + ar;
      if (gm < M) {
        if (A_IS_BF16) {
          const __bf16* src = (const __bf16*)Araw + (size_t)gm * K + kt + ac;
          #pragma unroll
          for (int j = 0; j < 16; j++) As[ar * LDA + ac + j] = src[j];
          if (kt + BK < K) __builtin_prefetch(src + BK);
        } else {
          const float* src = (const float*)Araw + (size_t)gm * K + kt + ac;
          #pragma unroll
          for (int j = 0; j < 16; j++) As[ar * LDA + ac + j] = (__bf16)src[j];
          if (kt + BK < K) __builtin_prefetch(src + BK);
        }
      } else {
        #pragma unroll
        for (int j = 0; j < 16; j++) As[ar * LDA + ac + j] = (__bf16)0.0f;
      }
    }
    // ---- stage B tile (weights, f32 -> bf16) ----
    {
      const int gn = nBase + br;
      if (gn < N) {
        const float* src = W + (size_t)gn * K + kt + bc;
        #pragma unroll
        for (int j = 0; j < 8; j++) Bs[br * LDB + bc + j] = (__bf16)src[j];
        if (kt + BK < K) __builtin_prefetch(src + BK);
      } else {
        #pragma unroll
        for (int j = 0; j < 8; j++) Bs[br * LDB + bc + j] = (__bf16)0.0f;
      }
    }
#if HAS_TDM
    if (A_IS_BF16 && wave == 0) __builtin_amdgcn_s_wait_tensorcnt(0);
#endif
    __syncthreads();

    // ---- load fragments per ISA 7.12.2 layouts and WMMA ----
    BF16Frag a[2], b[2];
    #pragma unroll
    for (int i = 0; i < 2; i++) {
      const int m = wm * 32 + i * 16 + l15;
      const unsigned int* rp = (const unsigned int*)&As[m * LDA];
      #pragma unroll
      for (int r = 0; r < 8; r++) {
        const int dw = (r & 3) + (lhalf ? 4 : 0) + ((r >= 4) ? 8 : 0);
        a[i].u[r] = rp[dw];
      }
    }
    #pragma unroll
    for (int j = 0; j < 2; j++) {
      const int n = wn * 32 + j * 16 + l15;
      const unsigned int* rp = (const unsigned int*)&Bs[n * LDB];
      #pragma unroll
      for (int r = 0; r < 8; r++) b[j].u[r] = rp[(r & 3) + (lhalf ? 4 : 0) + ((r >= 4) ? 8 : 0)];
    }
    #pragma unroll
    for (int i = 0; i < 2; i++)
      #pragma unroll
      for (int j = 0; j < 2; j++)
        acc[i][j].v = __builtin_amdgcn_wmma_f32_16x16x32_bf16(
            false, a[i].v, false, b[j].v, (short)0, acc[i][j].v, false, false);
  }

  // ---- epilogue: bias (+ReLU), store f32 or bf16 ----
  #pragma unroll
  for (int i = 0; i < 2; i++) {
    #pragma unroll
    for (int j = 0; j < 2; j++) {
      const int col = nBase + wn * 32 + j * 16 + l15;
      #pragma unroll
      for (int r = 0; r < 8; r++) {
        const int row = mBase + wm * 32 + i * 16 + (lhalf ? 8 + r : r);
        if (row < M && col < N) {
          float v = acc[i][j].f[r] + bias[col];
          if (RELU) v = fmaxf(v, 0.0f);
          if (OUT_BF16)
            ((__bf16*)Craw)[(size_t)row * N + col] = (__bf16)v;
          else
            ((float*)Craw)[(size_t)row * N + col] = v;
        }
      }
    }
  }
}

// ---------------------------------------------------------------------------
// Flash attention: grid (B*H, S/64), 128 threads = 4 waves, wave = 16 q rows.
// qkv is bf16 (B,S,3E); out is f32 (B,S,E).  scale folded into Q load.
// K/V tiles staged via GLOBAL_LOAD_ASYNC_TO_LDS_B128 when available.
// ---------------------------------------------------------------------------
__global__ __launch_bounds__(128) void flash_attn(
    const __bf16* __restrict__ qkv, float* __restrict__ out) {
  __shared__ __bf16 Qs[64 * 72];
  __shared__ __bf16 Ks[64 * 72];
  __shared__ __bf16 Vs[64 * 72];
  __shared__ __bf16 Ps[4][16 * 72];

  const int bh = blockIdx.x;
  const int b  = bh / NH, h = bh % NH;
  const int qb = blockIdx.y;
  const int tid = threadIdx.x, lane = tid & 31, wave = tid >> 5;
  const int lhalf = lane >> 4, l15 = lane & 15;
  const size_t rstride = 3 * NE;   // qkv elements per token

  // ---- stage Q (pre-scaled by 1/sqrt(64)) ----
  {
    const int r = tid >> 1, cb = (tid & 1) * 32;
    const int s = qb * 64 + r;
    const __bf16* src = qkv + (size_t)(b * NS + s) * rstride + h * 64 + cb;
    #pragma unroll
    for (int j = 0; j < 32; j++)
      Qs[r * 72 + cb + j] = (__bf16)((float)src[j] * 0.125f);
  }
  __syncthreads();

  // per-wave Q A-fragments (K-dim = head dim 64 -> two 32-chunks)
  BF16Frag aq0, aq1;
  {
    const int m = wave * 16 + l15;
    const unsigned int* rp = (const unsigned int*)&Qs[m * 72];
    #pragma unroll
    for (int r = 0; r < 8; r++) {
      const int dw = (r & 3) + (lhalf ? 4 : 0) + ((r >= 4) ? 8 : 0);
      aq0.u[r] = rp[dw];
      aq1.u[r] = rp[dw + 16];
    }
  }

  float rm[8], rl[8];
  F32Acc oacc[4];
  #pragma unroll
  for (int r = 0; r < 8; r++) { rm[r] = -1e30f; rl[r] = 0.0f; }
  #pragma unroll
  for (int d = 0; d < 4; d++)
    #pragma unroll
    for (int r = 0; r < 8; r++) oacc[d].f[r] = 0.0f;

  for (int kb = 0; kb < NS / 64; kb++) {
    __syncthreads();
    // ---- stage K, V tiles ----
    {
      const int r = tid >> 1, cb = (tid & 1) * 32;
      const int s = kb * 64 + r;
      const __bf16* kg = qkv + (size_t)(b * NS + s) * rstride + NE + h * 64 + cb;
      const __bf16* vg = kg + NE;            // V lives at +E relative to K
      __bf16* kl = &Ks[r * 72 + cb];
      __bf16* vl = &Vs[r * 72 + cb];
#if HAS_ASYNC_LDS
      // 64B per thread per tile: four b128 async copies, VGPR-free path.
      AS1 async_v4i* kgv = (AS1 async_v4i*)kg;
      AS1 async_v4i* vgv = (AS1 async_v4i*)vg;
      AS3 async_v4i* klv = (AS3 async_v4i*)kl;
      AS3 async_v4i* vlv = (AS3 async_v4i*)vl;
      __builtin_amdgcn_global_load_async_to_lds_b128(kgv, klv, 0, 0);
      __builtin_amdgcn_global_load_async_to_lds_b128(kgv, klv, 16, 0);
      __builtin_amdgcn_global_load_async_to_lds_b128(kgv, klv, 32, 0);
      __builtin_amdgcn_global_load_async_to_lds_b128(kgv, klv, 48, 0);
      __builtin_amdgcn_global_load_async_to_lds_b128(vgv, vlv, 0, 0);
      __builtin_amdgcn_global_load_async_to_lds_b128(vgv, vlv, 16, 0);
      __builtin_amdgcn_global_load_async_to_lds_b128(vgv, vlv, 32, 0);
      __builtin_amdgcn_global_load_async_to_lds_b128(vgv, vlv, 48, 0);
      __builtin_amdgcn_s_wait_asynccnt(0);
#else
      const unsigned int* ksrc = (const unsigned int*)kg;
      const unsigned int* vsrc = (const unsigned int*)vg;
      unsigned int* kd = (unsigned int*)kl;
      unsigned int* vd = (unsigned int*)vl;
      #pragma unroll
      for (int j = 0; j < 16; j++) { kd[j] = ksrc[j]; vd[j] = vsrc[j]; }
#endif
    }
    __syncthreads();

    // ---- S = Q K^T  (B-matrix = K tile: lane n = key, K-dim = d) ----
    F32Acc sf[4];
    #pragma unroll
    for (int nt = 0; nt < 4; nt++) {
      const int kn = nt * 16 + l15;
      const unsigned int* rp = (const unsigned int*)&Ks[kn * 72];
      BF16Frag bk0, bk1;
      #pragma unroll
      for (int r = 0; r < 8; r++) {
        bk0.u[r] = rp[(r & 3) + (lhalf ? 4 : 0) + ((r >= 4) ? 8 : 0)];
        bk1.u[r] = rp[(r & 3) + (lhalf ? 4 : 0) + ((r >= 4) ? 8 : 0) + 16];
      }
      F32Acc z;
      #pragma unroll
      for (int r = 0; r < 8; r++) z.f[r] = 0.0f;
      sf[nt].v = __builtin_amdgcn_wmma_f32_16x16x32_bf16(
          false, aq0.v, false, bk0.v, (short)0, z.v, false, false);
      sf[nt].v = __builtin_amdgcn_wmma_f32_16x16x32_bf16(
          false, aq1.v, false, bk1.v, (short)0, sf[nt].v, false, false);
    }

    // ---- online softmax (row stats across 16 lanes of each half) ----
    float nm[8], sc[8], ps[8];
    #pragma unroll
    for (int r = 0; r < 8; r++) {
      float t = fmaxf(fmaxf(sf[0].f[r], sf[1].f[r]),
                      fmaxf(sf[2].f[r], sf[3].f[r]));
      t = redMax16(t);
      nm[r] = fmaxf(rm[r], t);
      sc[r] = __expf(rm[r] - nm[r]);
      ps[r] = 0.0f;
    }
    #pragma unroll
    for (int nt = 0; nt < 4; nt++)
      #pragma unroll
      for (int r = 0; r < 8; r++) {
        const float p = __expf(sf[nt].f[r] - nm[r]);
        ps[r] += p;
        const int rowl = lhalf ? 8 + r : r;
        Ps[wave][rowl * 72 + nt * 16 + l15] = (__bf16)p;
      }
    #pragma unroll
    for (int r = 0; r < 8; r++) {
      rl[r] = rl[r] * sc[r] + redSum16(ps[r]);
      rm[r] = nm[r];
    }
    #pragma unroll
    for (int d = 0; d < 4; d++)
      #pragma unroll
      for (int r = 0; r < 8; r++) oacc[d].f[r] *= sc[r];
    __syncthreads();   // P staged (C-layout -> A-layout through LDS)

    // ---- O += P @ V ----
    BF16Frag ap0, ap1;
    {
      const unsigned int* rp = (const unsigned int*)&Ps[wave][l15 * 72];
      #pragma unroll
      for (int r = 0; r < 8; r++) {
        const int dw = (r & 3) + (lhalf ? 4 : 0) + ((r >= 4) ? 8 : 0);
        ap0.u[r] = rp[dw];
        ap1.u[r] = rp[dw + 16];
      }
    }
    #pragma unroll
    for (int dt = 0; dt < 4; dt++) {
      const int n = dt * 16 + l15;   // d column
      BF16Frag bv0, bv1;
      #pragma unroll
      for (int r = 0; r < 8; r++) {
        const int k0 = (lhalf ? 8 : 0) + 2 * (r & 3) + ((r >= 4) ? 16 : 0);
        unsigned int lo0 = *(const unsigned short*)&Vs[k0 * 72 + n];
        unsigned int hi0 = *(const unsigned short*)&Vs[(k0 + 1) * 72 + n];
        unsigned int lo1 = *(const unsigned short*)&Vs[(k0 + 32) * 72 + n];
        unsigned int hi1 = *(const unsigned short*)&Vs[(k0 + 33) * 72 + n];
        bv0.u[r] = lo0 | (hi0 << 16);
        bv1.u[r] = lo1 | (hi1 << 16);
      }
      oacc[dt].v = __builtin_amdgcn_wmma_f32_16x16x32_bf16(
          false, ap0.v, false, bv0.v, (short)0, oacc[dt].v, false, false);
      oacc[dt].v = __builtin_amdgcn_wmma_f32_16x16x32_bf16(
          false, ap1.v, false, bv1.v, (short)0, oacc[dt].v, false, false);
    }
  }

  // ---- epilogue: O / l -> out[b, s, h*64 + d] ----
  #pragma unroll
  for (int dt = 0; dt < 4; dt++)
    #pragma unroll
    for (int r = 0; r < 8; r++) {
      const int rowl = lhalf ? 8 + r : r;
      const int s = qb * 64 + wave * 16 + rowl;
      const int c = h * 64 + dt * 16 + l15;
      out[(size_t)(b * NS + s) * NE + c] = oacc[dt].f[r] / rl[r];
    }
}

// ---------------------------------------------------------------------------
// x = LayerNorm(x + hdelta) * g + b   (one wave per row of 512)
// ---------------------------------------------------------------------------
__global__ __launch_bounds__(256) void add_ln(
    float* __restrict__ x, const float* __restrict__ hd,
    const float* __restrict__ g, const float* __restrict__ bb) {
  const int lane = threadIdx.x & 31, wave = threadIdx.x >> 5;
  const int row = blockIdx.x * 8 + wave;
  const size_t base = (size_t)row * NE;
  float v[16];
  float s = 0.0f;
  #pragma unroll
  for (int i = 0; i < 16; i++) {
    const int c = i * 32 + lane;
    v[i] = x[base + c] + hd[base + c];
    s += v[i];
  }
  const float mean = redSum32(s) * (1.0f / NE);
  float s2 = 0.0f;
  #pragma unroll
  for (int i = 0; i < 16; i++) { const float d = v[i] - mean; s2 += d * d; }
  const float var = redSum32(s2) * (1.0f / NE);
  const float rstd = rsqrtf(var + 1e-5f);
  #pragma unroll
  for (int i = 0; i < 16; i++) {
    const int c = i * 32 + lane;
    x[base + c] = (v[i] - mean) * rstd * g[c] + bb[c];
  }
}

// ---------------------------------------------------------------------------
// CRF forward scan: single WG, thread (b,j) owns alpha[b][j]; 512 steps.
// ---------------------------------------------------------------------------
__global__ __launch_bounds__(384) void crf_forward(
    const float* __restrict__ feats, const float* __restrict__ trans,
    const int* __restrict__ seq_len, float* __restrict__ fwd) {
  __shared__ float alpha[NB][NT];
  __shared__ float tr[NT][NT];
  const int tid = threadIdx.x;
  const int b = tid / NT, j = tid % NT;
  if (tid < NT * NT) tr[tid / NT][tid % NT] = trans[tid];
  alpha[b][j] = (j == TAG_START) ? 0.0f : NEGV;
  __syncthreads();
  const int len = seq_len[b];
  for (int t = 0; t < NS; t++) {
    float m = -1e30f;
    #pragma unroll
    for (int i = 0; i < NT; i++) m = fmaxf(m, alpha[b][i] + tr[j][i]);
    float s = 0.0f;
    #pragma unroll
    for (int i = 0; i < NT; i++) s += __expf(alpha[b][i] + tr[j][i] - m);
    const float nv = feats[((size_t)b * NS + t) * NT + j] + m + __logf(s);
    const float res = (t < len) ? nv : alpha[b][j];
    __syncthreads();
    alpha[b][j] = res;
    __syncthreads();
  }
  if (j == 0) {
    float m = -1e30f;
    #pragma unroll
    for (int i = 0; i < NT; i++) m = fmaxf(m, alpha[b][i] + tr[TAG_STOP][i]);
    float s = 0.0f;
    #pragma unroll
    for (int i = 0; i < NT; i++) s += __expf(alpha[b][i] + tr[TAG_STOP][i] - m);
    fwd[b] = m + __logf(s);
  }
}

// ---------------------------------------------------------------------------
// Gold score + mean(fwd - gold) -> out[0]
// ---------------------------------------------------------------------------
__global__ __launch_bounds__(32) void crf_final(
    const float* __restrict__ feats, const float* __restrict__ trans,
    const int* __restrict__ tags, const int* __restrict__ seq_len,
    const float* __restrict__ fwd, float* __restrict__ outv) {
  const int b = threadIdx.x;
  const int len = seq_len[b];
  int prev = TAG_START;
  float ts = 0.0f, es = 0.0f;
  for (int s = 0; s < NS; s++) {
    if (s < len) {
      const int tg = tags[b * NS + s];
      ts += trans[tg * NT + prev];
      es += feats[((size_t)b * NS + s) * NT + tg];
      prev = tg;
    }
  }
  const float gold = ts + es + trans[TAG_STOP * NT + prev];
  float d = fwd[b] - gold;
  d = redSum32(d);
  if (b == 0) outv[0] = d * (1.0f / NB);
}

// ---------------------------------------------------------------------------
// Host-side orchestration
// ---------------------------------------------------------------------------
extern "C" void kernel_launch(void* const* d_in, const int* in_sizes, int n_in,
                              void* d_out, int out_size, void* d_ws,
                              size_t ws_size, hipStream_t stream) {
  (void)in_sizes; (void)n_in; (void)out_size; (void)ws_size;

  const int*   sent = (const int*)d_in[0];
  const int*   slen = (const int*)d_in[1];
  const int*   tags = (const int*)d_in[2];
  const float* emb  = (const float*)d_in[3];
  const float* w_in = (const float*)d_in[4];
  const float* b_in = (const float*)d_in[5];
  const float* w_o  = (const float*)d_in[6];
  const float* b_o  = (const float*)d_in[7];
  const float* l1g  = (const float*)d_in[8];
  const float* l1b  = (const float*)d_in[9];
  const float* l2g  = (const float*)d_in[10];
  const float* l2b  = (const float*)d_in[11];
  const float* f1w  = (const float*)d_in[12];
  const float* f1b  = (const float*)d_in[13];
  const float* f2w  = (const float*)d_in[14];
  const float* f2b  = (const float*)d_in[15];
  const float* tgw  = (const float*)d_in[16];
  const float* tgb  = (const float*)d_in[17];
  const float* trn  = (const float*)d_in[18];

  const int BS = NB * NS;                  // 16384 rows
  char* ws = (char*)d_ws;
  // region layout (reuse across phases):
  //   X  : fp32 activations           33,554,432 B
  //   T1 : bf16 qkv / bf16 ffn-hidden / f32 feats   67,108,864 B
  //   T2 : fp32 attn-out / ffn-out    33,554,432 B
  //   T3 : fp32 proj temp / fwd score 33,554,432 B   (total ~160 MB)
  float*   X  = (float*)ws;
  char*    T1 = ws + (size_t)33554432;
  float*   T2 = (float*)(ws + (size_t)33554432 + 67108864);
  float*   T3 = (float*)(ws + (size_t)33554432 + 67108864 + 33554432);
  __bf16*  QKV = (__bf16*)T1;
  __bf16*  HID = (__bf16*)T1;
  float*   FEATS = (float*)T1;
  float*   FWD = T3;

  const dim3 blk256(256), blk128(128);

  // 1) embedding
  embed_gather<<<BS, blk256, 0, stream>>>(sent, emb, X);

  for (int l = 0; l < NL; l++) {
    // 2) QKV projection -> bf16   (M=BS, N=3E, K=E)
    gemm_xwT<0, 1, 0><<<dim3((3 * NE) / BN, BS / BM), blk256, 0, stream>>>(
        X, w_in + (size_t)l * 3 * NE * NE, b_in + (size_t)l * 3 * NE, QKV,
        BS, 3 * NE, NE);
    // 3) flash attention -> T2 (f32)
    flash_attn<<<dim3(NB * NH, NS / 64), blk128, 0, stream>>>(QKV, T2);
    // 4) output projection -> T3 (f32)
    gemm_xwT<0, 0, 0><<<dim3(NE / BN, BS / BM), blk256, 0, stream>>>(
        T2, w_o + (size_t)l * NE * NE, b_o + (size_t)l * NE, T3, BS, NE, NE);
    // 5) x = LN(x + attn)
    add_ln<<<BS / 8, blk256, 0, stream>>>(X, T3, l1g + (size_t)l * NE,
                                          l1b + (size_t)l * NE);
    // 6) FFN1 + ReLU -> bf16 hidden
    gemm_xwT<0, 1, 1><<<dim3(NDF / BN, BS / BM), blk256, 0, stream>>>(
        X, f1w + (size_t)l * NDF * NE, f1b + (size_t)l * NDF, HID,
        BS, NDF, NE);
    // 7) FFN2 -> T2 (f32), A staged by TDM when available
    gemm_xwT<1, 0, 0><<<dim3(NE / BN, BS / BM), blk256, 0, stream>>>(
        HID, f2w + (size_t)l * NE * NDF, f2b + (size_t)l * NE, T2,
        BS, NE, NDF);
    // 8) x = LN(x + ffn)
    add_ln<<<BS / 8, blk256, 0, stream>>>(X, T2, l2g + (size_t)l * NE,
                                          l2b + (size_t)l * NE);
  }

  // 9) tag projection (N=12, bounds-guarded single n-tile)
  gemm_xwT<0, 0, 0><<<dim3(1, BS / BM), blk256, 0, stream>>>(
      X, tgw, tgb, FEATS, BS, NT, NE);

  // 10) CRF forward scan
  crf_forward<<<1, dim3(NB * NT), 0, stream>>>(FEATS, trn, slen, FWD);

  // 11) gold score + mean
  crf_final<<<1, dim3(NB), 0, stream>>>(FEATS, trn, tags, slen, FWD,
                                        (float*)d_out);
}
